// DVAE_DeepGMG_58205396795647
// MI455X (gfx1250) — compile-verified, matrix operand order checked by reading, not run on hardware
//
#include <hip/hip_runtime.h>
#include <math.h>

// ---------------------------------------------------------------------------
// DVAE_DeepGMG encoder for MI455X (gfx1250), wave32 + v_wmma_f32_16x16x32_bf16
// One workgroup per graph; H state lives in LDS across all 3 GRU steps.
// ---------------------------------------------------------------------------

typedef __bf16 bf16_t;
typedef __attribute__((ext_vector_type(16))) __bf16 v16bf;
typedef __attribute__((ext_vector_type(8)))  float  v8f;

#define BATCH 512
#define NNODE 32
#define HSD   512
#define GSD   1024   // 2*HS
#define NZD   56
#define TED   3

union FragU {
  v16bf  v;
  float4 q[2];
  bf16_t e[16];
};

// Fragment load from row-major bf16 storage (row stride ld elements).
// Serves both A-matrices (rows = M) and B-matrices stored N-major (rows = N).
// Per ISA 7.12.2 (16-bit 16x32 A): lane m=(l&15), g=(l>>4);
//   elems 0..7  -> K = kbase + g*8 + 0..7      (one 16B load)
//   elems 8..15 -> K = kbase + 16 + g*8 + 0..7 (one 16B load)
__device__ __forceinline__ v16bf frag_rm(const bf16_t* src, int row_base, int ld,
                                         int kbase, int lane) {
  const int r = row_base + (lane & 15);
  const int g = lane >> 4;
  const bf16_t* p = src + (size_t)r * ld + kbase + g * 8;
  FragU f;
  f.q[0] = *(const float4*)(p);
  f.q[1] = *(const float4*)(p + 16);
  return f.v;
}

// B fragment gathered from K-major (row-major [K][N]) storage (slow path,
// used only for the tiny P = adjT @ H GEMM, <1% of FLOPs).
__device__ __forceinline__ v16bf frag_b_gather(const bf16_t* src, int ld,
                                               int kbase, int nbase, int lane) {
  const int n = nbase + (lane & 15);
  const int g = lane >> 4;
  FragU f;
#pragma unroll
  for (int i = 0; i < 8; ++i)
    f.e[i] = src[(size_t)(kbase + g * 8 + i) * ld + n];
#pragma unroll
  for (int i = 0; i < 8; ++i)
    f.e[8 + i] = src[(size_t)(kbase + 16 + g * 8 + i) * ld + n];
  return f.v;
}

__device__ __forceinline__ v8f wmma_bf16(v16bf a, v16bf b, v8f c) {
  // (neg_a, A, neg_b, B, c_mod, C, reuse_a, reuse_b)
  return __builtin_amdgcn_wmma_f32_16x16x32_bf16(false, a, false, b, (short)0, c,
                                                 false, false);
}

// Store a 16x16 f32 accumulator tile as bf16 into row-major [M][N] storage.
// Per ISA 7.12.2: VGPR r holds row (r + 8*(lane>>4)), col = lane&15.
__device__ __forceinline__ void store_tile_rm(bf16_t* dst, int ld, int mbase,
                                              int nbase, int lane, v8f acc) {
  const int n = nbase + (lane & 15);
  const int g = lane >> 4;
#pragma unroll
  for (int r = 0; r < 8; ++r)
    dst[(size_t)(mbase + r + 8 * g) * ld + n] = (bf16_t)acc[r];
}

__device__ __forceinline__ float sigm(float x) { return 1.f / (1.f + expf(-x)); }

// ---------------------------------------------------------------------------
// Small helper kernels
// ---------------------------------------------------------------------------

__global__ void k_conv_direct(const float* __restrict__ src,
                              bf16_t* __restrict__ dst, int n) {
  int i = blockIdx.x * 256 + threadIdx.x;
  if (i < n) dst[i] = (bf16_t)src[i];
}

// dst (C rows x R cols, row-major) = transpose of src (R rows x C cols)
__global__ void k_conv_transpose(const float* __restrict__ src,
                                 bf16_t* __restrict__ dst, int R, int C) {
  int i = blockIdx.x * 256 + threadIdx.x;
  if (i < R * C) {
    int r = i / C, c = i % C;
    dst[(size_t)c * R + r] = (bf16_t)src[(size_t)r * C + c];
  }
}

__global__ void k_zero_u32(unsigned int* __restrict__ p) {
  if (threadIdx.x == 0) p[0] = 0u;
}

// has_pred[v] = any_{b,u} adj[b,u,v] > 0 -> 32-bit mask.
// v is the innermost dim of adj, so (flat index & 31) == v.
__global__ void k_has_pred(const float* __restrict__ adj,
                           unsigned int* __restrict__ mask) {
  unsigned int local = 0u;
  const size_t total = (size_t)BATCH * NNODE * NNODE;
  const size_t stride = (size_t)gridDim.x * blockDim.x;
  for (size_t i = (size_t)blockIdx.x * blockDim.x + threadIdx.x; i < total;
       i += stride)
    if (adj[i] > 0.f) local |= 1u << (i & 31);
  if (local) atomicOr(mask, local);
}

// mu/logvar heads, full fp32: out (2, B, NZ)
__global__ void k_fc(const float* __restrict__ Hg, const float* __restrict__ w1,
                     const float* __restrict__ b1, const float* __restrict__ w2,
                     const float* __restrict__ b2, float* __restrict__ out) {
  int b = blockIdx.x;
  int t = threadIdx.x;  // 128 threads, 112 active
  if (t >= 2 * NZD) return;
  int which = t / NZD, z = t % NZD;
  const float* w = which ? w2 : w1;
  float acc = which ? b2[z] : b1[z];
  const float* hg = Hg + (size_t)b * GSD;
  for (int k = 0; k < GSD; ++k) acc += hg[k] * w[(size_t)k * NZD + z];
  out[(size_t)which * BATCH * NZD + (size_t)b * NZD + z] = acc;
}

// ---------------------------------------------------------------------------
// Fused per-graph kernel: init + 3 GRU message-passing steps + gated readout
// ---------------------------------------------------------------------------

__global__ __launch_bounds__(256) void k_fused(
    const int* __restrict__ node_types, const float* __restrict__ adj,
    const float* __restrict__ finit_w, const float* __restrict__ finit_b,
    const float* __restrict__ fe_w /* row 512 = we */,
    const float* __restrict__ fe_b, const bf16_t* __restrict__ WnT,
    const bf16_t* __restrict__ WsT, const bf16_t* __restrict__ wih_bf,
    const bf16_t* __restrict__ whh_bf, const float* __restrict__ bih,
    const float* __restrict__ bhh, const bf16_t* __restrict__ gateT,
    const bf16_t* __restrict__ mapT, const float* __restrict__ gate_b,
    const unsigned int* __restrict__ haspp, float* __restrict__ Hg) {
  __shared__ __align__(16) bf16_t sX[2][NNODE * HSD];   // H (cur) / P & H(next)
  __shared__ __align__(16) bf16_t sAv[NNODE * GSD];     // Av (32 x 1024)
  __shared__ __align__(16) bf16_t sAdjT[NNODE * NNODE]; // adj^T (32 x 32)
  __shared__ float sDeg[NNODE];

  const int b = blockIdx.x;
  const int tid = threadIdx.x;
  const int lane = tid & 31;
  const int w = tid >> 5;  // 8 waves
  const unsigned int mask = haspp[0];
  const float* we = fe_w + (size_t)HSD * GSD;  // fe_w[512, :]

  const v8f zf = {0.f, 0.f, 0.f, 0.f, 0.f, 0.f, 0.f, 0.f};

  // ---- init: H[n,k] = finit_w[type(n), k] + finit_b[k] ----
  for (int i = tid; i < NNODE * HSD; i += 256) {
    int nrow = i >> 9, k = i & (HSD - 1);
    int ty = node_types[b * NNODE + nrow];
    sX[0][i] = (bf16_t)(finit_w[(size_t)ty * HSD + k] + finit_b[k]);
  }
  // ---- adj^T (bf16, exact for 0/1 values) ----
  for (int i = tid; i < NNODE * NNODE; i += 256) {
    int u = i >> 5, v = i & 31;
    sAdjT[v * NNODE + u] = (bf16_t)adj[(size_t)b * NNODE * NNODE + u * NNODE + v];
  }
  // ---- deg[v] = sum_u adj[b,u,v] ----
  if (tid < NNODE) {
    float s = 0.f;
    for (int u = 0; u < NNODE; ++u)
      s += adj[(size_t)b * NNODE * NNODE + u * NNODE + tid];
    sDeg[tid] = s;
  }
  __syncthreads();

  int cur = 0;
  for (int t = 0; t < TED; ++t) {
    const bf16_t* Hc = sX[cur];
    bf16_t* Pb = sX[cur ^ 1];  // P lives where H(next) will be written

    // ---- P = adjT @ H   (32 x 512, K = 32 -> single WMMA per tile) ----
    for (int ct = w; ct < HSD / 16; ct += 8) {
      int nb = ct * 16;
      v16bf bb = frag_b_gather(Hc, HSD, 0, nb, lane);
#pragma unroll
      for (int mt = 0; mt < 2; ++mt) {
        v16bf a = frag_rm(sAdjT, mt * 16, NNODE, 0, lane);
        v8f d = wmma_bf16(a, bb, zf);
        store_tile_rm(Pb, HSD, mt * 16, nb, lane, d);
      }
    }
    __syncthreads();

    // ---- Av = P@Wn + deg (.) (H@Ws + we + fe_b)   (32 x 1024) ----
    for (int et = w; et < GSD / 16; et += 8) {
      int nb = et * 16;
      v8f t2a = zf, t2b = zf, ava = zf, avb = zf;
      const bf16_t* wsRow = WsT + (size_t)(nb + (lane & 15)) * HSD;
      const bf16_t* wnRow = WnT + (size_t)(nb + (lane & 15)) * HSD;
#pragma unroll 2
      for (int kb = 0; kb < HSD; kb += 32) {
        if (kb + 32 < HSD) {  // hide L2 latency of next weight chunk
          __builtin_prefetch(wsRow + kb + 32, 0, 3);
          __builtin_prefetch(wnRow + kb + 32, 0, 3);
        }
        v16bf h0 = frag_rm(Hc, 0, HSD, kb, lane);
        v16bf h1 = frag_rm(Hc, 16, HSD, kb, lane);
        v16bf p0 = frag_rm(Pb, 0, HSD, kb, lane);
        v16bf p1 = frag_rm(Pb, 16, HSD, kb, lane);
        v16bf bws = frag_rm(WsT, nb, HSD, kb, lane);
        v16bf bwn = frag_rm(WnT, nb, HSD, kb, lane);
        t2a = wmma_bf16(h0, bws, t2a);
        t2b = wmma_bf16(h1, bws, t2b);
        ava = wmma_bf16(p0, bwn, ava);
        avb = wmma_bf16(p1, bwn, avb);
      }
      int col = nb + (lane & 15);
      int gg = lane >> 4;
      float addc = we[col] + fe_b[col];
#pragma unroll
      for (int r = 0; r < 8; ++r) {
        int m0 = r + 8 * gg, m1 = 16 + r + 8 * gg;
        sAv[(size_t)m0 * GSD + col] = (bf16_t)(ava[r] + sDeg[m0] * (t2a[r] + addc));
        sAv[(size_t)m1 * GSD + col] = (bf16_t)(avb[r] + sDeg[m1] * (t2b[r] + addc));
      }
    }
    __syncthreads();

    // ---- gates: gx = Av@wih^T, gh = H@whh^T, GRU update -> H(next) ----
    const bf16_t* wih_t = wih_bf + (size_t)t * 3 * HSD * GSD;   // (1536, 1024)
    const bf16_t* whh_t = whh_bf + (size_t)t * 3 * HSD * HSD;   // (1536, 512)
    const float* bih_t = bih + t * 3 * HSD;
    const float* bhh_t = bhh + t * 3 * HSD;
    bf16_t* Hn = Pb;  // overwrite dead P

    for (int ct = w; ct < HSD / 16; ct += 8) {
      int cb = ct * 16;
      v8f gx[2][3], gh[2][3];
#pragma unroll
      for (int mt = 0; mt < 2; ++mt)
#pragma unroll
        for (int gi = 0; gi < 3; ++gi) { gx[mt][gi] = zf; gh[mt][gi] = zf; }

      const bf16_t* wihRow = wih_t + (size_t)(cb + (lane & 15)) * GSD;
#pragma unroll 2
      for (int kb = 0; kb < GSD; kb += 32) {  // K over Av (1024)
        if (kb + 32 < GSD) {
          __builtin_prefetch(wihRow + kb + 32, 0, 3);
          __builtin_prefetch(wihRow + (size_t)HSD * GSD + kb + 32, 0, 3);
          __builtin_prefetch(wihRow + (size_t)2 * HSD * GSD + kb + 32, 0, 3);
        }
        v16bf a0 = frag_rm(sAv, 0, GSD, kb, lane);
        v16bf a1 = frag_rm(sAv, 16, GSD, kb, lane);
#pragma unroll
        for (int gi = 0; gi < 3; ++gi) {
          v16bf bb = frag_rm(wih_t, cb + gi * HSD, GSD, kb, lane);
          gx[0][gi] = wmma_bf16(a0, bb, gx[0][gi]);
          gx[1][gi] = wmma_bf16(a1, bb, gx[1][gi]);
        }
      }
      const bf16_t* whhRow = whh_t + (size_t)(cb + (lane & 15)) * HSD;
#pragma unroll 2
      for (int kb = 0; kb < HSD; kb += 32) {  // K over H (512)
        if (kb + 32 < HSD) {
          __builtin_prefetch(whhRow + kb + 32, 0, 3);
          __builtin_prefetch(whhRow + (size_t)HSD * HSD + kb + 32, 0, 3);
          __builtin_prefetch(whhRow + (size_t)2 * HSD * HSD + kb + 32, 0, 3);
        }
        v16bf a0 = frag_rm(Hc, 0, HSD, kb, lane);
        v16bf a1 = frag_rm(Hc, 16, HSD, kb, lane);
#pragma unroll
        for (int gi = 0; gi < 3; ++gi) {
          v16bf bb = frag_rm(whh_t, cb + gi * HSD, HSD, kb, lane);
          gh[0][gi] = wmma_bf16(a0, bb, gh[0][gi]);
          gh[1][gi] = wmma_bf16(a1, bb, gh[1][gi]);
        }
      }

      int col = cb + (lane & 15);
      int gg = lane >> 4;
      float bi0 = bih_t[col], bi1 = bih_t[col + HSD], bi2 = bih_t[col + 2 * HSD];
      float bh0 = bhh_t[col], bh1 = bhh_t[col + HSD], bh2 = bhh_t[col + 2 * HSD];
#pragma unroll
      for (int mt = 0; mt < 2; ++mt) {
#pragma unroll
        for (int r = 0; r < 8; ++r) {
          int m = mt * 16 + r + 8 * gg;
          float hold = (float)Hc[(size_t)m * HSD + col];
          float rr = sigm(gx[mt][0][r] + bi0 + gh[mt][0][r] + bh0);
          float zz = sigm(gx[mt][1][r] + bi1 + gh[mt][1][r] + bh1);
          float nn = tanhf(gx[mt][2][r] + bi2 + rr * (gh[mt][2][r] + bh2));
          float hnew = (1.f - zz) * nn + zz * hold;
          if (!((mask >> m) & 1u)) hnew = hold;
          Hn[(size_t)m * HSD + col] = (bf16_t)hnew;
        }
      }
    }
    __syncthreads();
    cur ^= 1;
  }

  // ---- readout: Hg[b,j] = sum_v sigmoid((H@gate_w)[v,j]+gate_b[j]) * (H@mapper_w)[v,j]
  const bf16_t* Hf = sX[cur];
  for (int jt = w; jt < GSD / 16; jt += 8) {
    int nb = jt * 16;
    v8f ag0 = zf, ag1 = zf, am0 = zf, am1 = zf;
    const bf16_t* gRow = gateT + (size_t)(nb + (lane & 15)) * HSD;
    const bf16_t* mRow = mapT + (size_t)(nb + (lane & 15)) * HSD;
#pragma unroll 2
    for (int kb = 0; kb < HSD; kb += 32) {
      if (kb + 32 < HSD) {
        __builtin_prefetch(gRow + kb + 32, 0, 3);
        __builtin_prefetch(mRow + kb + 32, 0, 3);
      }
      v16bf a0 = frag_rm(Hf, 0, HSD, kb, lane);
      v16bf a1 = frag_rm(Hf, 16, HSD, kb, lane);
      v16bf bg = frag_rm(gateT, nb, HSD, kb, lane);
      v16bf bm = frag_rm(mapT, nb, HSD, kb, lane);
      ag0 = wmma_bf16(a0, bg, ag0);
      ag1 = wmma_bf16(a1, bg, ag1);
      am0 = wmma_bf16(a0, bm, am0);
      am1 = wmma_bf16(a1, bm, am1);
    }
    int col = nb + (lane & 15);
    float gb = gate_b[col];
    float part = 0.f;
#pragma unroll
    for (int r = 0; r < 8; ++r) part += sigm(ag0[r] + gb) * am0[r];
#pragma unroll
    for (int r = 0; r < 8; ++r) part += sigm(ag1[r] + gb) * am1[r];
    part += __shfl_xor(part, 16, 32);  // combine lanes l and l+16 (rows split)
    if (lane < 16) Hg[(size_t)b * GSD + col] = part;
  }
}

// ---------------------------------------------------------------------------
// Launch
// ---------------------------------------------------------------------------

extern "C" void kernel_launch(void* const* d_in, const int* in_sizes, int n_in,
                              void* d_out, int out_size, void* d_ws,
                              size_t ws_size, hipStream_t stream) {
  (void)in_sizes; (void)n_in; (void)out_size; (void)ws_size;

  const int*   node_types = (const int*)d_in[0];
  const float* adj        = (const float*)d_in[1];
  const float* finit_w    = (const float*)d_in[2];
  const float* finit_b    = (const float*)d_in[3];
  const float* fe_w       = (const float*)d_in[4];
  const float* fe_b       = (const float*)d_in[5];
  const float* grue_wih   = (const float*)d_in[6];
  const float* grue_whh   = (const float*)d_in[7];
  const float* grue_bih   = (const float*)d_in[8];
  const float* grue_bhh   = (const float*)d_in[9];
  const float* gate_w     = (const float*)d_in[10];
  const float* gate_b     = (const float*)d_in[11];
  const float* mapper_w   = (const float*)d_in[12];
  const float* fc1_w      = (const float*)d_in[13];
  const float* fc1_b      = (const float*)d_in[14];
  const float* fc2_w      = (const float*)d_in[15];
  const float* fc2_b      = (const float*)d_in[16];
  float* out = (float*)d_out;

  char* ws = (char*)d_ws;
  size_t off = 0;
  auto carve = [&](size_t bytes) -> char* {
    char* p = ws + off;
    off += (bytes + 255) & ~(size_t)255;
    return p;
  };

  bf16_t* wih_bf = (bf16_t*)carve((size_t)3 * 1536 * 1024 * 2);
  bf16_t* whh_bf = (bf16_t*)carve((size_t)3 * 1536 * 512 * 2);
  bf16_t* WnT    = (bf16_t*)carve((size_t)1024 * 512 * 2);
  bf16_t* WsT    = (bf16_t*)carve((size_t)1024 * 512 * 2);
  bf16_t* gateT  = (bf16_t*)carve((size_t)1024 * 512 * 2);
  bf16_t* mapT   = (bf16_t*)carve((size_t)1024 * 512 * 2);
  float*  Hg     = (float*)carve((size_t)512 * 1024 * 4);
  unsigned int* hasp = (unsigned int*)carve(256);

  const int n1 = 3 * 1536 * 1024;
  k_conv_direct<<<(n1 + 255) / 256, 256, 0, stream>>>(grue_wih, wih_bf, n1);
  const int n2 = 3 * 1536 * 512;
  k_conv_direct<<<(n2 + 255) / 256, 256, 0, stream>>>(grue_whh, whh_bf, n2);
  const int nt = 512 * 1024;
  k_conv_transpose<<<(nt + 255) / 256, 256, 0, stream>>>(fe_w, WnT, 512, 1024);
  k_conv_transpose<<<(nt + 255) / 256, 256, 0, stream>>>(fe_w + (size_t)513 * 1024,
                                                         WsT, 512, 1024);
  k_conv_transpose<<<(nt + 255) / 256, 256, 0, stream>>>(gate_w, gateT, 512, 1024);
  k_conv_transpose<<<(nt + 255) / 256, 256, 0, stream>>>(mapper_w, mapT, 512, 1024);
  k_zero_u32<<<1, 32, 0, stream>>>(hasp);
  k_has_pred<<<64, 256, 0, stream>>>(adj, hasp);

  k_fused<<<BATCH, 256, 0, stream>>>(node_types, adj, finit_w, finit_b, fe_w,
                                     fe_b, WnT, WsT, wih_bf, whh_bf, grue_bih,
                                     grue_bhh, gateT, mapT, gate_b, hasp, Hg);

  k_fc<<<BATCH, 128, 0, stream>>>(Hg, fc1_w, fc1_b, fc2_w, fc2_b, out);
}
